// MultiHeadAttention2_88046829568789
// MI455X (gfx1250) — matrix operand order repaired and checked by reading, not compiled, not run on hardware
//
#include <hip/hip_runtime.h>
#include <hip/hip_bf16.h>
#include <stdint.h>

#define B_    4
#define S_    2048
#define H_    16
#define HID_  1024
#define DHEAD 64

typedef __attribute__((ext_vector_type(16))) __bf16 v16bf;
typedef __attribute__((ext_vector_type(8)))  float  v8f;

union Frag16 { v16bf v; unsigned int u[8]; };
union Pack8  { uint4 q; unsigned short s[8]; };

__device__ __forceinline__ unsigned short f2bf(float f) {
  unsigned int u = __float_as_uint(f);
  unsigned int r = u + 0x7FFFu + ((u >> 16) & 1u);   // round-to-nearest-even
  return (unsigned short)(r >> 16);
}
__device__ __forceinline__ v8f wmma_bf16(v16bf a, v16bf b, v8f c) {
  return __builtin_amdgcn_wmma_f32_16x16x32_bf16(false, a, false, b, (short)0, c, false, false);
}
// Async 16B global->LDS copy (GLOBAL_LOAD_ASYNC_TO_LDS_B128, tracked by ASYNCcnt)
__device__ __forceinline__ void async_b128(unsigned ldsoff, const void* gaddr) {
  asm volatile("global_load_async_to_lds_b128 %0, %1, off"
               :: "v"(ldsoff), "v"((unsigned long long)(size_t)gaddr) : "memory");
}

// ---------------------------------------------------------------------------
// One-shot fp32 -> bf16 conversion (8 elems/thread).
// ---------------------------------------------------------------------------
__global__ __launch_bounds__(256) void f32_to_bf16_kernel(
    const float* __restrict__ in, unsigned short* __restrict__ out)
{
  size_t idx = ((size_t)blockIdx.x * 256 + threadIdx.x) * 8;
  float4 a = *(const float4*)(in + idx);
  float4 b = *(const float4*)(in + idx + 4);
  Pack8 u;
  u.s[0] = f2bf(a.x); u.s[1] = f2bf(a.y); u.s[2] = f2bf(a.z); u.s[3] = f2bf(a.w);
  u.s[4] = f2bf(b.x); u.s[5] = f2bf(b.y); u.s[6] = f2bf(b.z); u.s[7] = f2bf(b.w);
  *(uint4*)(out + idx) = u.q;
}

// ---------------------------------------------------------------------------
// One-shot fp32 W[K][N] -> bf16 Wt[N][K] (convert + transpose via LDS tile).
// 64x64 tiles, grid (N/64, K/64); both global sides coalesced.
// ---------------------------------------------------------------------------
__global__ __launch_bounds__(256) void convT_kernel(
    const float* __restrict__ in, unsigned short* __restrict__ out, int K, int N)
{
  __shared__ __align__(16) unsigned short tile[64][72];   // [n][k], 144B rows
  const int t = threadIdx.x;
  const int n0 = blockIdx.x * 64, k0 = blockIdx.y * 64;
  #pragma unroll
  for (int i = 0; i < 16; i++) {
    int id = t + 256 * i; int kk = id >> 6, nn = id & 63;
    tile[nn][kk] = f2bf(in[(size_t)(k0 + kk) * N + n0 + nn]);
  }
  __syncthreads();
  #pragma unroll
  for (int i = 0; i < 2; i++) {
    int id = t + 256 * i; int nn = id >> 3, k8 = (id & 7) * 8;
    *(uint4*)(out + (size_t)(n0 + nn) * K + k0 + k8) = *(uint4*)&tile[nn][k8];
  }
}

// ---------------------------------------------------------------------------
// GEMM: Y = (X_bf16[M,K] @ Wt_bf16[N,K]^T + bias) * scale (+ residual).
// 128x128 block tile, K-step 32, double-buffered LDS filled entirely by
// global_load_async_to_lds_b128 (4 ops/thread/tile). 8 waves as 4(M)x2(N),
// each wave 32x64 -> 8 WMMA per K-step.
// VTRANS: write output as [b, col, s] (i.e. V^T layout for attention).
// ---------------------------------------------------------------------------
template <bool BF16OUT, bool VTRANS>
__global__ __launch_bounds__(256) void gemm_bf16_kernel(
    const unsigned short* __restrict__ X, const unsigned short* __restrict__ Wt,
    const float* __restrict__ bias, const float* __restrict__ res,
    void* __restrict__ Yout, float scale, int M, int N, int K)
{
  __shared__ __align__(16) unsigned short lds_a[2][128][40];  // [m][k]
  __shared__ __align__(16) unsigned short lds_w[2][128][40];  // [n][k]

  const int t = threadIdx.x;
  const int lane = t & 31, wave = t >> 5;
  const int lanelo = lane & 15, halfsel = lane >> 4;
  const int waveM = wave & 3, waveN = wave >> 2;
  const int m0 = blockIdx.y * 128, n0 = blockIdx.x * 128;

  // staging: thread owns two 16B chunks per tile per matrix
  const int cr = t >> 2, c8 = (t & 3) * 8;
  const unsigned short* pa = X  + (size_t)(m0 + cr) * K + c8;
  const unsigned short* pw = Wt + (size_t)(n0 + cr) * K + c8;
  unsigned adst[2][2], wdst[2][2];
  #pragma unroll
  for (int bi = 0; bi < 2; bi++)
    #pragma unroll
    for (int j = 0; j < 2; j++) {
      adst[bi][j] = (unsigned)(size_t)&lds_a[bi][cr + 64 * j][c8];
      wdst[bi][j] = (unsigned)(size_t)&lds_w[bi][cr + 64 * j][c8];
    }

  const v8f zero = {0.f,0.f,0.f,0.f,0.f,0.f,0.f,0.f};
  v8f acc[2][4];
  #pragma unroll
  for (int ma = 0; ma < 2; ma++)
    #pragma unroll
    for (int nb = 0; nb < 4; nb++) acc[ma][nb] = zero;

  const int NSTEP = K / 32;
  // prologue: tile 0 -> buffer 0
  #pragma unroll
  for (int j = 0; j < 2; j++) {
    async_b128(adst[0][j], pa + (size_t)64 * j * K);
    async_b128(wdst[0][j], pw + (size_t)64 * j * K);
  }

  for (int step = 0; step < NSTEP; step++) {
    __syncthreads();                 // all waves done reading buf[(step-1)&1]
    if (step + 1 < NSTEP) {
      int nbuf = (step + 1) & 1;
      size_t koff = (size_t)(step + 1) * 32;
      #pragma unroll
      for (int j = 0; j < 2; j++) {
        async_b128(adst[nbuf][j], pa + koff + (size_t)64 * j * K);
        async_b128(wdst[nbuf][j], pw + koff + (size_t)64 * j * K);
      }
      asm volatile("s_wait_asynccnt 0x4" ::: "memory");  // tile(step) done (in-order)
    } else {
      asm volatile("s_wait_asynccnt 0x0" ::: "memory");
    }
    __syncthreads();                 // tile(step) visible to all waves
    const int cb = step & 1;

    Frag16 a[2];
    #pragma unroll
    for (int ma = 0; ma < 2; ma++) {
      int arow = waveM * 32 + ma * 16 + lanelo;
      #pragma unroll
      for (int v = 0; v < 8; v++) {
        int kidx = ((v < 4) ? 0 : 16) + 8 * halfsel + 2 * (v & 3);
        a[ma].u[v] = *(const unsigned int*)&lds_a[cb][arow][kidx];
      }
    }
    #pragma unroll
    for (int nb = 0; nb < 4; nb++) {
      Frag16 bf;
      int ncol = waveN * 64 + nb * 16 + lanelo;
      #pragma unroll
      for (int v = 0; v < 8; v++)
        bf.u[v] = *(const unsigned int*)&lds_w[cb][ncol][((v < 4) ? 0 : 16) + 8 * halfsel + 2 * (v & 3)];
      acc[0][nb] = wmma_bf16(a[0].v, bf.v, acc[0][nb]);
      acc[1][nb] = wmma_bf16(a[1].v, bf.v, acc[1][nb]);
    }
  }

  #pragma unroll
  for (int ma = 0; ma < 2; ma++) {
    #pragma unroll
    for (int nb = 0; nb < 4; nb++) {
      #pragma unroll
      for (int v = 0; v < 8; v++) {
        int row = m0 + waveM * 32 + ma * 16 + v + 8 * halfsel;
        int col = n0 + waveN * 64 + nb * 16 + lanelo;
        float val = (acc[ma][nb][v] + bias[col]) * scale;
        if (res) val += res[(size_t)row * N + col];
        if constexpr (VTRANS) {
          // [b, col, s] layout: (b*HID + col)*S + s
          size_t idx = ((size_t)((row >> 11) * HID_ + col)) * (size_t)S_ + (row & (S_ - 1));
          ((unsigned short*)Yout)[idx] = f2bf(val);
        } else if constexpr (BF16OUT) {
          ((unsigned short*)Yout)[(size_t)row * N + col] = f2bf(val);
        } else {
          ((float*)Yout)[(size_t)row * N + col] = val;
        }
      }
    }
  }
}

// ---------------------------------------------------------------------------
// Flash attention, all-bf16 operands (1/sqrt(dk) folded into Q projection).
// Block = (b, h, 128 q-rows); 8 waves x 16 q-rows. Double-buffered 32-key
// chunks: K [key][d] and V^T [d][key] both staged with async b128 copies.
// ---------------------------------------------------------------------------
__global__ __launch_bounds__(256) void attention_kernel(
    const unsigned short* __restrict__ Yq, const unsigned short* __restrict__ Yk,
    const unsigned short* __restrict__ Yvt, const unsigned char* __restrict__ mask,
    unsigned short* __restrict__ ctx)
{
  __shared__ __align__(16) unsigned short kt[2][32][72];   // [key][d]
  __shared__ __align__(16) unsigned short vt[2][64][40];   // [d][key]
  __shared__ __align__(16) unsigned short pbuf[8][16][34]; // per-wave P [m][key]

  const int t = threadIdx.x;
  const int lane = t & 31, wave = t >> 5;
  const int lanelo = lane & 15, halfsel = lane >> 4;
  const int qx = blockIdx.x, h = blockIdx.y, b = blockIdx.z;

  // Q fragments (bf16 direct; pairs are dword loads)
  Frag16 qa0, qa1;
  {
    const int qrow = qx * 128 + wave * 16 + lanelo;
    const unsigned short* qp = Yq + ((size_t)(b * S_ + qrow)) * HID_ + h * DHEAD;
    #pragma unroll
    for (int v = 0; v < 8; v++) {
      int kidx = ((v < 4) ? 0 : 16) + 8 * halfsel + 2 * (v & 3);
      qa0.u[v] = *(const unsigned int*)(qp + kidx);
      qa1.u[v] = *(const unsigned int*)(qp + 32 + kidx);
    }
  }

  const v8f zero = {0.f,0.f,0.f,0.f,0.f,0.f,0.f,0.f};
  v8f acc[4] = {zero, zero, zero, zero};
  float mrow[8], lrow[8];
  #pragma unroll
  for (int v = 0; v < 8; v++) { mrow[v] = -1e30f; lrow[v] = 0.f; }

  // staging coords: K chunk = 32 rows x 128B; V chunk = 64 rows x 64B
  const int ks = t >> 3, kd8 = (t & 7) * 8;
  const int vd = t >> 2, vk8 = (t & 3) * 8;
  const unsigned short* kglob = Yk  + ((size_t)(b * S_ + ks)) * HID_ + h * DHEAD + kd8;
  const unsigned short* vglob = Yvt + ((size_t)(b * HID_ + h * DHEAD + vd)) * S_ + vk8;
  unsigned kdst[2], vdst[2];
  #pragma unroll
  for (int bi = 0; bi < 2; bi++) {
    kdst[bi] = (unsigned)(size_t)&kt[bi][ks][kd8];
    vdst[bi] = (unsigned)(size_t)&vt[bi][vd][vk8];
  }

  const int NCH = S_ / 32;
  async_b128(kdst[0], kglob);
  async_b128(vdst[0], vglob);

  for (int ci = 0; ci < NCH; ci++) {
    const int kc = ci * 32;
    __syncthreads();
    if (ci + 1 < NCH) {
      int nbuf = (ci + 1) & 1;
      async_b128(kdst[nbuf], kglob + (size_t)(kc + 32) * HID_);
      async_b128(vdst[nbuf], vglob + (kc + 32));
      asm volatile("s_wait_asynccnt 0x2" ::: "memory");
    } else {
      asm volatile("s_wait_asynccnt 0x0" ::: "memory");
    }
    __syncthreads();
    const int cb = ci & 1;

    // scores: 2 key-subtiles x (contraction 64 = 2 WMMA)
    v8f s[2];
    #pragma unroll
    for (int st = 0; st < 2; st++) {
      Frag16 b0, b1;
      int key = st * 16 + lanelo;
      #pragma unroll
      for (int v = 0; v < 8; v++) {
        int d = ((v < 4) ? 0 : 16) + 8 * halfsel + 2 * (v & 3);
        b0.u[v] = *(const unsigned int*)&kt[cb][key][d];
        b1.u[v] = *(const unsigned int*)&kt[cb][key][32 + d];
      }
      v8f p0 = wmma_bf16(qa0.v, b0.v, zero);
      s[st]  = wmma_bf16(qa1.v, b1.v, p0);
    }

    // mask (reference: where(mask, -1e9, s))
    #pragma unroll
    for (int st = 0; st < 2; st++) {
      #pragma unroll
      for (int v = 0; v < 8; v++) {
        int qrow = qx * 128 + wave * 16 + v + 8 * halfsel;
        int key  = kc + st * 16 + lanelo;
        if (mask[((size_t)(b * S_ + qrow)) * S_ + key]) s[st][v] = -1e9f;
      }
    }

    // online softmax; rows live in 16-lane halves -> xor 1/2/4/8
    #pragma unroll
    for (int v = 0; v < 8; v++) {
      float mx = fmaxf(s[0][v], s[1][v]);
      mx = fmaxf(mx, __shfl_xor(mx, 1));
      mx = fmaxf(mx, __shfl_xor(mx, 2));
      mx = fmaxf(mx, __shfl_xor(mx, 4));
      mx = fmaxf(mx, __shfl_xor(mx, 8));
      float mnew = fmaxf(mrow[v], mx);
      float corr = __expf(mrow[v] - mnew);
      float p0 = __expf(s[0][v] - mnew);
      float p1 = __expf(s[1][v] - mnew);
      float rs = p0 + p1;
      rs += __shfl_xor(rs, 1); rs += __shfl_xor(rs, 2);
      rs += __shfl_xor(rs, 4); rs += __shfl_xor(rs, 8);
      lrow[v] = lrow[v] * corr + rs;
      mrow[v] = mnew;
      #pragma unroll
      for (int nb = 0; nb < 4; nb++) acc[nb][v] *= corr;
      int m = v + 8 * halfsel;
      pbuf[wave][m][lanelo]      = f2bf(p0);
      pbuf[wave][m][16 + lanelo] = f2bf(p1);
    }
    asm volatile("s_wait_dscnt 0x0" ::: "memory");  // same-wave P store->load

    Frag16 pa;
    #pragma unroll
    for (int v = 0; v < 8; v++) {
      int kidx = ((v < 4) ? 0 : 16) + 8 * halfsel + 2 * (v & 3);
      pa.u[v] = *(const unsigned int*)&pbuf[wave][lanelo][kidx];
    }
    #pragma unroll
    for (int nb = 0; nb < 4; nb++) {
      Frag16 vb;
      int d = nb * 16 + lanelo;
      #pragma unroll
      for (int v = 0; v < 8; v++)
        vb.u[v] = *(const unsigned int*)&vt[cb][d][((v < 4) ? 0 : 16) + 8 * halfsel + 2 * (v & 3)];
      acc[nb] = wmma_bf16(pa.v, vb.v, acc[nb]);
    }
  }

  #pragma unroll
  for (int v = 0; v < 8; v++) {
    float inv = 1.f / lrow[v];
    int qrow = qx * 128 + wave * 16 + v + 8 * halfsel;
    #pragma unroll
    for (int nb = 0; nb < 4; nb++) {
      int col = h * DHEAD + nb * 16 + lanelo;
      ctx[((size_t)(b * S_ + qrow)) * HID_ + col] = f2bf(acc[nb][v] * inv);
    }
  }
}

// ---------------------------------------------------------------------------
// LayerNorm over last dim (1024), eps 1e-5. One block per row.
// ---------------------------------------------------------------------------
__global__ __launch_bounds__(256) void layernorm_kernel(
    const float* __restrict__ X, const float* __restrict__ g,
    const float* __restrict__ be, float* __restrict__ out)
{
  __shared__ float red[256];
  __shared__ float red2[256];
  const int row = blockIdx.x;
  const int t = threadIdx.x;
  const float* xp = X + (size_t)row * HID_;
  float vals[4], s = 0.f, sq = 0.f;
  #pragma unroll
  for (int i = 0; i < 4; i++) {
    float v = xp[t + 256 * i];
    vals[i] = v; s += v; sq += v * v;
  }
  red[t] = s; red2[t] = sq;
  __syncthreads();
  for (int off = 128; off > 0; off >>= 1) {
    if (t < off) { red[t] += red[t + off]; red2[t] += red2[t + off]; }
    __syncthreads();
  }
  float mean = red[0] * (1.f / HID_);
  float var  = red2[0] * (1.f / HID_) - mean * mean;
  float rstd = rsqrtf(var + 1e-5f);
  #pragma unroll
  for (int i = 0; i < 4; i++) {
    int c = t + 256 * i;
    out[(size_t)row * HID_ + c] = (vals[i] - mean) * rstd * g[c] + be[c];
  }
}

extern "C" void kernel_launch(void* const* d_in, const int* in_sizes, int n_in,
                              void* d_out, int out_size, void* d_ws, size_t ws_size,
                              hipStream_t stream)
{
  (void)in_sizes; (void)n_in; (void)out_size; (void)ws_size;
  const float* Q  = (const float*)d_in[0];
  const float* K  = (const float*)d_in[1];
  const float* V  = (const float*)d_in[2];
  const unsigned char* mask = (const unsigned char*)d_in[3];
  const float* Wq = (const float*)d_in[4];
  const float* bq = (const float*)d_in[5];
  const float* Wk = (const float*)d_in[6];
  const float* bk = (const float*)d_in[7];
  const float* Wv = (const float*)d_in[8];
  const float* bv = (const float*)d_in[9];
  const float* Wo = (const float*)d_in[10];
  const float* bo = (const float*)d_in[11];
  const float* lg = (const float*)d_in[12];
  const float* lb = (const float*)d_in[13];

  const size_t INE = (size_t)B_ * S_ * HID_;   // 8,388,608 elems
  const size_t WE  = (size_t)HID_ * HID_;      // 1,048,576 elems

  unsigned short* p = (unsigned short*)d_ws;
  unsigned short* Qb  = p; p += INE;
  unsigned short* Kb  = p; p += INE;
  unsigned short* Vb  = p; p += INE;
  unsigned short* Wqt = p; p += WE;
  unsigned short* Wkt = p; p += WE;
  unsigned short* Wvt = p; p += WE;
  unsigned short* Wot = p; p += WE;
  unsigned short* Yqb = p; p += INE;
  unsigned short* Ykb = p; p += INE;
  unsigned short* Yvt = p; p += INE;   // [b, hid, s] transposed layout
  unsigned short* Ctx = p; p += INE;
  float* Pre = (float*)Qb;             // Qb+Kb (32 MB) dead after projections

  dim3 blk(256);
  f32_to_bf16_kernel<<<dim3((unsigned)(INE / 2048)), blk, 0, stream>>>(Q, Qb);
  f32_to_bf16_kernel<<<dim3((unsigned)(INE / 2048)), blk, 0, stream>>>(K, Kb);
  f32_to_bf16_kernel<<<dim3((unsigned)(INE / 2048)), blk, 0, stream>>>(V, Vb);
  dim3 ctGrid(HID_ / 64, HID_ / 64);
  convT_kernel<<<ctGrid, blk, 0, stream>>>(Wq, Wqt, HID_, HID_);
  convT_kernel<<<ctGrid, blk, 0, stream>>>(Wk, Wkt, HID_, HID_);
  convT_kernel<<<ctGrid, blk, 0, stream>>>(Wv, Wvt, HID_, HID_);
  convT_kernel<<<ctGrid, blk, 0, stream>>>(Wo, Wot, HID_, HID_);

  dim3 gemmGrid(HID_ / 128, (B_ * S_) / 128);   // (8, 64)
  gemm_bf16_kernel<true , false><<<gemmGrid, blk, 0, stream>>>(Qb, Wqt, bq, nullptr, Yqb, 0.125f, B_ * S_, HID_, HID_);
  gemm_bf16_kernel<true , false><<<gemmGrid, blk, 0, stream>>>(Kb, Wkt, bk, nullptr, Ykb, 1.0f,   B_ * S_, HID_, HID_);
  gemm_bf16_kernel<true , true ><<<gemmGrid, blk, 0, stream>>>(Vb, Wvt, bv, nullptr, Yvt, 1.0f,   B_ * S_, HID_, HID_);

  dim3 attnGrid(S_ / 128, H_, B_);              // (16, 16, 4)
  attention_kernel<<<attnGrid, blk, 0, stream>>>(Yqb, Ykb, Yvt, mask, Ctx);

  gemm_bf16_kernel<false, false><<<gemmGrid, blk, 0, stream>>>(Ctx, Wot, bo, Q, Pre, 1.0f, B_ * S_, HID_, HID_);
  layernorm_kernel<<<dim3(B_ * S_), blk, 0, stream>>>(Pre, lg, lb, (float*)d_out);
}
